// GRU_NLP_WTA_15324443312626
// MI455X (gfx1250) — matrix operand order repaired and checked by reading, not compiled, not run on hardware
//
#include <hip/hip_runtime.h>
#include <hip/hip_bf16.h>
#include <math.h>

// ---------------------------------------------------------------------------
// Problem constants (from reference)
// ---------------------------------------------------------------------------
#define S_LEN 128
#define BATCH 32
#define IDIM  512
#define HDIM  512
#define CDIM  2048
#define VDIM  32000
#define NIND  307
#define ROWS  (S_LEN * BATCH)   // 4096

typedef _Float16 hvec8  __attribute__((ext_vector_type(8)));
typedef _Float16 hvec16 __attribute__((ext_vector_type(16)));
typedef float    fvec8  __attribute__((ext_vector_type(8)));

// ---------------------------------------------------------------------------
// WMMA fragment loaders (CDNA5 wave32 layouts, 16x16x32 f16)
//  A (16x32 MxK): lanes 0-15 -> row=lane, halves {K0..7, K16..23}
//                 lanes 16-31 -> row=lane-16, halves {K8..15, K24..31}
//  B (32x16 KxN): lane -> column = lane&15, same K halves as A:
//                 lanes<16 -> {K0..7, K16..23}, lanes>=16 -> {K8..15, K24..31}
// ---------------------------------------------------------------------------
__device__ __forceinline__ hvec16 frag_a(const _Float16* __restrict__ row0,
                                         int ld, int kt) {
  int l  = threadIdx.x & 31;
  int r  = l & 15;
  int ko = ((l >> 4) & 1) * 8 + kt;
  const hvec8* p0 = (const hvec8*)(row0 + (size_t)r * ld + ko);
  const hvec8* p1 = (const hvec8*)(row0 + (size_t)r * ld + ko + 16);
  hvec8 c0 = *p0, c1 = *p1;
  return __builtin_shufflevector(c0, c1, 0,1,2,3,4,5,6,7,8,9,10,11,12,13,14,15);
}

__device__ __forceinline__ hvec16 frag_b(const _Float16* __restrict__ row0,
                                         int ld, int kt) {
  int l  = threadIdx.x & 31;
  int n  = l & 15;
  int ko = ((l >> 4) & 1) * 8 + kt;
  const hvec8* p0 = (const hvec8*)(row0 + (size_t)n * ld + ko);
  const hvec8* p1 = (const hvec8*)(row0 + (size_t)n * ld + ko + 16);
  hvec8 c0 = *p0, c1 = *p1;
  return __builtin_shufflevector(c0, c1, 0,1,2,3,4,5,6,7,8,9,10,11,12,13,14,15);
}

__device__ __forceinline__ fvec8 wmma_f16(hvec16 a, hvec16 b, fvec8 c) {
  return __builtin_amdgcn_wmma_f32_16x16x32_f16(false, a, false, b,
                                                (short)0, c, false, false);
}

// CDNA5 async copy: global -> LDS, 16B per lane, tracked by ASYNCcnt.
// IOFFSET applies to BOTH the LDS and global addresses (ISA 10.7/15.18).
__device__ __forceinline__ void async_b128(unsigned lds_addr,
                                           unsigned long long gaddr) {
  asm volatile("global_load_async_to_lds_b128 %0, %1, off"
               :: "v"(lds_addr), "v"(gaddr) : "memory");
}
__device__ __forceinline__ void async_b128_off16(unsigned lds_addr,
                                                 unsigned long long gaddr) {
  asm volatile("global_load_async_to_lds_b128 %0, %1, off offset:16"
               :: "v"(lds_addr), "v"(gaddr) : "memory");
}
__device__ __forceinline__ void wait_async0() {
  asm volatile("s_wait_asynccnt 0x0" ::: "memory");
}
__device__ __forceinline__ unsigned lds_addr32(const void* p) {
  // LDS aperture: generic addr[31:0] == LDS byte address
  return (unsigned)(unsigned long long)p;
}

// ---------------------------------------------------------------------------
// Generic tiled GEMM:  C[M,N] = A[M,K] * B[N,K]^T + bias[N]
// A,B row-major f16; C f32.  M,N multiples of 128; K multiple of 32.
// Block 128x128x32, 256 threads (8 waves, 2x4 wave grid).
// Double-buffered async global->LDS staging overlapped with WMMA.
// ---------------------------------------------------------------------------
#define BM 128
#define BN 128
#define BK 32
#define LDSH 40   // padded halves per LDS row (80B, keeps 16B alignment)

__global__ __launch_bounds__(256) void gemm_f16_wmma(
    const _Float16* __restrict__ A, const _Float16* __restrict__ B,
    const float* __restrict__ bias, float* __restrict__ C,
    int M, int N, int K) {
  __shared__ _Float16 sA[2][BM * LDSH];
  __shared__ _Float16 sB[2][BN * LDSH];

  const int tid = threadIdx.x;
  const int n0  = blockIdx.x * BN;
  const int m0  = blockIdx.y * BM;

  const int lrow = tid >> 1;            // 0..127
  const int loff = (tid & 1) * 16;      // 0 or 16 halves

  const int wid  = tid >> 5;            // 0..7
  const int lane = tid & 31;
  const int wm   = wid & 1;             // 2 M-slabs of 64
  const int wn   = wid >> 1;            // 4 N-slabs of 32

  // per-thread staging addresses
  unsigned ldsA[2], ldsB[2];
  ldsA[0] = lds_addr32(&sA[0][lrow * LDSH + loff]);
  ldsA[1] = lds_addr32(&sA[1][lrow * LDSH + loff]);
  ldsB[0] = lds_addr32(&sB[0][lrow * LDSH + loff]);
  ldsB[1] = lds_addr32(&sB[1][lrow * LDSH + loff]);
  const _Float16* gA = A + (size_t)(m0 + lrow) * K + loff;
  const _Float16* gB = B + (size_t)(n0 + lrow) * K + loff;

  fvec8 acc[4][2];
#pragma unroll
  for (int i = 0; i < 4; ++i)
#pragma unroll
    for (int j = 0; j < 2; ++j) acc[i][j] = (fvec8)0.f;

  const int nk = K / BK;

  // prologue: stage tile 0 into buffer 0
  {
    unsigned long long ga = (unsigned long long)gA;
    unsigned long long gb = (unsigned long long)gB;
    async_b128(ldsA[0], ga);  async_b128_off16(ldsA[0], ga);
    async_b128(ldsB[0], gb);  async_b128_off16(ldsB[0], gb);
  }

  for (int i = 0; i < nk; ++i) {
    const int cur = i & 1;
    wait_async0();        // my async loads for buffer `cur` are in LDS
    __syncthreads();      // all waves: cur staged AND done reading nxt

    if (i + 1 < nk) {     // overlap next tile's loads with this tile's WMMAs
      const int kt = (i + 1) * BK;
      unsigned long long ga = (unsigned long long)(gA + kt);
      unsigned long long gb = (unsigned long long)(gB + kt);
      async_b128(ldsA[cur ^ 1], ga);  async_b128_off16(ldsA[cur ^ 1], ga);
      async_b128(ldsB[cur ^ 1], gb);  async_b128_off16(ldsB[cur ^ 1], gb);
    }

    hvec16 af[4], bf[2];
#pragma unroll
    for (int mt = 0; mt < 4; ++mt)
      af[mt] = frag_a(&sA[cur][(wm * 64 + mt * 16) * LDSH], LDSH, 0);
#pragma unroll
    for (int nt = 0; nt < 2; ++nt)
      bf[nt] = frag_b(&sB[cur][(wn * 32 + nt * 16) * LDSH], LDSH, 0);

#pragma unroll
    for (int mt = 0; mt < 4; ++mt)
#pragma unroll
      for (int nt = 0; nt < 2; ++nt)
        acc[mt][nt] = wmma_f16(af[mt], bf[nt], acc[mt][nt]);
  }

  // epilogue: C layout row = r + 8*(lane>=16), col = lane&15
  const int col0 = n0 + wn * 32;
  const int row0 = m0 + wm * 64;
  const int roff = ((lane >> 4) & 1) * 8;
  const int cl   = lane & 15;
#pragma unroll
  for (int mt = 0; mt < 4; ++mt) {
#pragma unroll
    for (int nt = 0; nt < 2; ++nt) {
      int c = col0 + nt * 16 + cl;
      float bv = bias[c];
      int rb = row0 + mt * 16 + roff;
#pragma unroll
      for (int r = 0; r < 8; ++r)
        C[(size_t)(rb + r) * N + c] = acc[mt][nt][r] + bv;
    }
  }
}

// ---------------------------------------------------------------------------
// GRU step: gh = h_prev * w_hh^T (+b_hh), fused gates, h update.
// Grid: 8 blocks x 128 threads (4 waves). Block owns 64 columns of H;
// wave owns 16 columns, computes r/z/n slabs (3 gates x 2 M-tiles).
// ---------------------------------------------------------------------------
__global__ __launch_bounds__(128) void gru_step_kernel(
    const _Float16* __restrict__ hprev,   // [32,512] f16
    const _Float16* __restrict__ whh,     // [1536,512] f16
    const float* __restrict__ bhh,        // [1536]
    const float* __restrict__ gx_s,       // [32, 1536] (this step's slice)
    float* __restrict__ hf32,             // [32,512] running f32 state
    _Float16* __restrict__ hout_s) {      // [32,512] f16 (this step's output)
  const int tid  = threadIdx.x;
  const int wid  = tid >> 5;
  const int lane = tid & 31;
  const int jw   = blockIdx.x * 64 + wid * 16;   // column base for this wave

  fvec8 acc[2][3];
#pragma unroll
  for (int mt = 0; mt < 2; ++mt)
#pragma unroll
    for (int g = 0; g < 3; ++g) acc[mt][g] = (fvec8)0.f;

  for (int kt = 0; kt < HDIM; kt += 32) {
    hvec16 a0 = frag_a(hprev,             HDIM, kt);   // rows 0..15
    hvec16 a1 = frag_a(hprev + 16 * HDIM, HDIM, kt);   // rows 16..31
#pragma unroll
    for (int g = 0; g < 3; ++g) {
      hvec16 b = frag_b(whh + (size_t)(g * HDIM + jw) * HDIM, HDIM, kt);
      acc[0][g] = wmma_f16(a0, b, acc[0][g]);
      acc[1][g] = wmma_f16(a1, b, acc[1][g]);
    }
  }

  const int roff = ((lane >> 4) & 1) * 8;
  const int j    = jw + (lane & 15);
  const float bhr = bhh[j], bhz = bhh[HDIM + j], bhn = bhh[2 * HDIM + j];
#pragma unroll
  for (int mt = 0; mt < 2; ++mt) {
#pragma unroll
    for (int r = 0; r < 8; ++r) {
      int b = mt * 16 + roff + r;
      const float* gxr = gx_s + (size_t)b * (3 * HDIM);
      float ghr = acc[mt][0][r] + bhr;
      float ghz = acc[mt][1][r] + bhz;
      float ghn = acc[mt][2][r] + bhn;
      float rg = 1.f / (1.f + __expf(-(gxr[j] + ghr)));
      float zg = 1.f / (1.f + __expf(-(gxr[HDIM + j] + ghz)));
      float ng = tanhf(gxr[2 * HDIM + j] + rg * ghn);
      float hold = hf32[(size_t)b * HDIM + j];
      float hnew = (1.f - zg) * ng + zg * hold;
      hf32[(size_t)b * HDIM + j] = hnew;
      hout_s[(size_t)b * HDIM + j] = (_Float16)hnew;
    }
  }
}

// ---------------------------------------------------------------------------
// WTA top-NIND + L2 normalize per row of hc [4096 x 2048] -> f16
// Radix-select (4x8-bit MSB-first) on order-preserving float->uint keys.
// ---------------------------------------------------------------------------
__device__ __forceinline__ unsigned f2key(float f) {
  unsigned u = __float_as_uint(f);
  return (u & 0x80000000u) ? ~u : (u | 0x80000000u);
}

__global__ __launch_bounds__(256) void wta_norm_kernel(
    const float* __restrict__ hc, _Float16* __restrict__ hcn) {
  __shared__ float    vals[CDIM];
  __shared__ unsigned hist[256];
  __shared__ unsigned s_sel[2];
  __shared__ float    s_red[256];
  const int t = threadIdx.x;
  const size_t base = (size_t)blockIdx.x * CDIM;

  for (int i = t; i < CDIM; i += 256) vals[i] = hc[base + i];
  __syncthreads();

  unsigned prefix = 0, pmask = 0, k = NIND;
  for (int shift = 24; shift >= 0; shift -= 8) {
    hist[t] = 0;
    __syncthreads();
    for (int i = t; i < CDIM; i += 256) {
      unsigned key = f2key(vals[i]);
      if ((key & pmask) == prefix)
        atomicAdd(&hist[(key >> shift) & 255u], 1u);
    }
    __syncthreads();
    if (t == 0) {
      unsigned cum = 0;
      int d = 255;
      for (;; --d) {
        unsigned c = hist[d];
        if (cum + c >= k || d == 0) { s_sel[0] = (unsigned)d; s_sel[1] = k - cum; break; }
        cum += c;
      }
    }
    __syncthreads();
    prefix |= (s_sel[0] << shift);
    pmask  |= (255u << shift);
    k = s_sel[1];
  }
  const unsigned T = prefix;   // exact k-th largest key

  float ss = 0.f;
  for (int i = t; i < CDIM; i += 256) {
    float v = vals[i];
    if (f2key(v) >= T) ss += v * v;
  }
  s_red[t] = ss;
  __syncthreads();
#pragma unroll
  for (int o = 128; o > 0; o >>= 1) {
    if (t < o) s_red[t] += s_red[t + o];
    __syncthreads();
  }
  const float scale = rsqrtf(s_red[0]);

  for (int i = t; i < CDIM; i += 256) {
    float v = vals[i];
    float o = (f2key(v) >= T) ? v * scale : 0.f;
    hcn[base + i] = (_Float16)o;
  }
}

// ---------------------------------------------------------------------------
// In-place log_softmax along rows of [4096 x V] (one online pass + write pass)
// ---------------------------------------------------------------------------
__global__ __launch_bounds__(256) void log_softmax_kernel(
    float* __restrict__ out, int V) {
  __shared__ float sm[256], ssum[256];
  const int t = threadIdx.x;
  const size_t base = (size_t)blockIdx.x * V;

  float m = -3.4e38f, s = 0.f;
  for (int i = t; i < V; i += 256) {
    float v = out[base + i];
    if (v > m) { s = s * __expf(m - v) + 1.f; m = v; }
    else       { s += __expf(v - m); }
  }
  sm[t] = m; ssum[t] = s;
  __syncthreads();
#pragma unroll
  for (int o = 128; o > 0; o >>= 1) {
    if (t < o) {
      float m2 = sm[t + o], s2 = ssum[t + o];
      float M = fmaxf(sm[t], m2);
      ssum[t] = ssum[t] * __expf(sm[t] - M) + s2 * __expf(m2 - M);
      sm[t] = M;
    }
    __syncthreads();
  }
  const float lse = sm[0] + __logf(ssum[0]);
  for (int i = t; i < V; i += 256) out[base + i] -= lse;
}

// ---------------------------------------------------------------------------
// Utility kernels
// ---------------------------------------------------------------------------
__global__ void cvt_f32_f16_kernel(const float* __restrict__ in,
                                   _Float16* __restrict__ out, size_t n) {
  size_t i = (size_t)blockIdx.x * blockDim.x + threadIdx.x;
  size_t stride = (size_t)gridDim.x * blockDim.x;
  for (; i < n; i += stride) out[i] = (_Float16)in[i];
}

__global__ void copy_f32_kernel(const float* __restrict__ in,
                                float* __restrict__ out, size_t n) {
  size_t i = (size_t)blockIdx.x * blockDim.x + threadIdx.x;
  size_t stride = (size_t)gridDim.x * blockDim.x;
  for (; i < n; i += stride) out[i] = in[i];
}

// ---------------------------------------------------------------------------
// Host orchestration
// ---------------------------------------------------------------------------
extern "C" void kernel_launch(void* const* d_in, const int* in_sizes, int n_in,
                              void* d_out, int out_size, void* d_ws, size_t ws_size,
                              hipStream_t stream) {
  const float* x     = (const float*)d_in[0];   // [S,B,I]
  const float* h0    = (const float*)d_in[1];   // [1,B,H]
  const float* w_ih  = (const float*)d_in[2];   // [3H,I]
  const float* w_hh  = (const float*)d_in[3];   // [3H,H]
  const float* b_ih  = (const float*)d_in[4];   // [3H]
  const float* b_hh  = (const float*)d_in[5];   // [3H]
  const float* w_h2c = (const float*)d_in[6];   // [C,H]
  const float* b_h2c = (const float*)d_in[7];   // [C]
  const float* w_c2o = (const float*)d_in[8];   // [V,C]
  const float* b_c2o = (const float*)d_in[9];   // [V]

  char* w = (char*)d_ws;
  size_t off = 0;
  auto alloc = [&](size_t bytes) { size_t r = off; off += (bytes + 255) & ~(size_t)255; return r; };
  _Float16* x16     = (_Float16*)(w + alloc((size_t)ROWS * IDIM * 2));
  _Float16* wih16   = (_Float16*)(w + alloc((size_t)3 * HDIM * IDIM * 2));
  _Float16* whh16   = (_Float16*)(w + alloc((size_t)3 * HDIM * HDIM * 2));
  _Float16* wh2c16  = (_Float16*)(w + alloc((size_t)CDIM * HDIM * 2));
  _Float16* wc2o16  = (_Float16*)(w + alloc((size_t)VDIM * CDIM * 2));
  float*    gx      = (float*)   (w + alloc((size_t)ROWS * 3 * HDIM * 4));
  _Float16* hout16  = (_Float16*)(w + alloc((size_t)ROWS * HDIM * 2));
  _Float16* hinit16 = (_Float16*)(w + alloc((size_t)BATCH * HDIM * 2));
  float*    hf32    = (float*)   (w + alloc((size_t)BATCH * HDIM * 4));
  float*    hc      = (float*)   (w + alloc((size_t)ROWS * CDIM * 4));
  _Float16* hcn16   = (_Float16*)(w + alloc((size_t)ROWS * CDIM * 2));

  float* out = (float*)d_out;                      // [S,B,V] then [1,B,H]

  // f32 -> f16 conversions
  cvt_f32_f16_kernel<<<512, 256, 0, stream>>>(x, x16, (size_t)ROWS * IDIM);
  cvt_f32_f16_kernel<<<512, 256, 0, stream>>>(w_ih, wih16, (size_t)3 * HDIM * IDIM);
  cvt_f32_f16_kernel<<<512, 256, 0, stream>>>(w_hh, whh16, (size_t)3 * HDIM * HDIM);
  cvt_f32_f16_kernel<<<512, 256, 0, stream>>>(w_h2c, wh2c16, (size_t)CDIM * HDIM);
  cvt_f32_f16_kernel<<<4096, 256, 0, stream>>>(w_c2o, wc2o16, (size_t)VDIM * CDIM);
  cvt_f32_f16_kernel<<<16, 256, 0, stream>>>(h0, hinit16, (size_t)BATCH * HDIM);
  copy_f32_kernel<<<16, 256, 0, stream>>>(h0, hf32, (size_t)BATCH * HDIM);

  // gx = x * w_ih^T + b_ih   [4096 x 1536], K=512
  gemm_f16_wmma<<<dim3(3 * HDIM / BN, ROWS / BM), 256, 0, stream>>>(
      x16, wih16, b_ih, gx, ROWS, 3 * HDIM, IDIM);

  // GRU recurrence: 128 sequential fused steps
  for (int s = 0; s < S_LEN; ++s) {
    const _Float16* hprev = (s == 0) ? hinit16 : (hout16 + (size_t)(s - 1) * BATCH * HDIM);
    gru_step_kernel<<<HDIM / 64, 128, 0, stream>>>(
        hprev, whh16, b_hh, gx + (size_t)s * BATCH * 3 * HDIM,
        hf32, hout16 + (size_t)s * BATCH * HDIM);
  }

  // hc = hout * w_h2c^T + b_h2c   [4096 x 2048], K=512
  gemm_f16_wmma<<<dim3(CDIM / BN, ROWS / BM), 256, 0, stream>>>(
      hout16, wh2c16, b_h2c, hc, ROWS, CDIM, HDIM);

  // WTA top-307 + L2 normalize -> f16
  wta_norm_kernel<<<ROWS, 256, 0, stream>>>(hc, hcn16);

  // logits = hcn * w_c2o^T + b_c2o   [4096 x 32000], K=2048 -> d_out
  gemm_f16_wmma<<<dim3(VDIM / BN, ROWS / BM), 256, 0, stream>>>(
      hcn16, wc2o16, b_c2o, out, ROWS, VDIM, CDIM);

  // in-place log_softmax on d_out
  log_softmax_kernel<<<ROWS, 256, 0, stream>>>(out, VDIM);

  // hn -> tail of d_out
  copy_f32_kernel<<<16, 256, 0, stream>>>(hf32, out + (size_t)ROWS * VDIM,
                                          (size_t)BATCH * HDIM);
}
